// LSTM_7404523618677
// MI455X (gfx1250) — compile-verified, matrix operand order checked by reading, not run on hardware
//
#include <hip/hip_runtime.h>
#include <hip/hip_bf16.h>

// ---------------------------------------------------------------------------
// CDNA5 (gfx1250) TreeLSTM: bf16 WMMA GEMMs + async global->LDS staging +
// bf16 mirrors of recurrent state + device-side wavefront schedule.
// wave32; v_wmma_f32_16x16x32_bf16 with f32 accumulators.
// ---------------------------------------------------------------------------

typedef __attribute__((ext_vector_type(16))) __bf16 v16bf;
typedef __attribute__((ext_vector_type(8)))  __bf16 v8bf;
typedef __attribute__((ext_vector_type(8)))  float  v8f;

#define TL_ITERS 24
#define TL_K 512

__device__ inline int imin_i(int a, int b) { return a < b ? a : b; }

__device__ inline unsigned short f32_to_bf16_bits(float f) {
    unsigned u = __builtin_bit_cast(unsigned, f);
    unsigned r = u + 0x7FFFu + ((u >> 16) & 1u);   // round-to-nearest-even
    return (unsigned short)(r >> 16);
}

__device__ inline float sigmoidf_dev(float x) {
    return 1.0f / (1.0f + __expf(-x));
}

// CDNA5 async global->LDS copy (ASYNCcnt-tracked, no VGPR round trip).
// GVS mode: mem = SADDR(64b sgpr pair) + VADDR(32b byte offset); dst = LDS
// byte address in a VGPR (low 32 bits of the generic shared pointer).
__device__ inline void async_copy_b64(const void* gbase, unsigned byte_off,
                                      void* lds_dst) {
    unsigned lds_addr = (unsigned)(unsigned long long)lds_dst;
    asm volatile("global_load_async_to_lds_b64 %0, %1, %2"
                 :
                 : "v"(lds_addr), "v"(byte_off),
                   "s"((unsigned long long)gbase)
                 : "memory");
}
__device__ inline void wait_async0() {
    asm volatile("s_wait_asynccnt 0x0" ::: "memory");
}

// 16x32 bf16 fragment from a bf16 row (A) or weight row (B; row-major by
// out-col). Lane l: row/col = l&15; half = l>>4 selects K {half*8..} and
// {16+half*8..}: two b128 loads, zero conversion ALU.
__device__ inline v16bf load_frag16(const __bf16* rowptr, int k0, int kh) {
    const v8bf* p0 = reinterpret_cast<const v8bf*>(rowptr + k0 + kh * 8);
    const v8bf* p1 = reinterpret_cast<const v8bf*>(rowptr + k0 + 16 + kh * 8);
    v8bf lo = *p0, hi = *p1;
    v16bf v;
#pragma unroll
    for (int i = 0; i < 8; ++i) { v[i] = lo[i]; v[8 + i] = hi[i]; }
    return v;
}

__device__ inline v8f wmma_bf16(v16bf a, v16bf b, v8f c) {
    return __builtin_amdgcn_wmma_f32_16x16x32_bf16(false, a, false, b, (short)0, c,
                                                   false, false);
}

// ---------------------------------------------------------------------------
// Utility kernels
// ---------------------------------------------------------------------------

__global__ void tl_zero_u32(unsigned* p, long long n) {
    long long i = (long long)blockIdx.x * blockDim.x + threadIdx.x;
    long long stride = (long long)gridDim.x * blockDim.x;
    for (; i < n; i += stride) p[i] = 0u;
}

__global__ void tl_cvt_bf16(const float* src, unsigned short* dst, long long n) {
    long long i = (long long)blockIdx.x * blockDim.x + threadIdx.x;
    long long stride = (long long)gridDim.x * blockDim.x;
    for (; i < n; i += stride) dst[i] = f32_to_bf16_bits(src[i]);
}

__global__ void tl_copy_f32(const float* src, float* dst, long long n) {
    long long i = (long long)blockIdx.x * blockDim.x + threadIdx.x;
    long long stride = (long long)gridDim.x * blockDim.x;
    for (; i < n; i += stride) dst[i] = src[i];
}

__global__ void tl_count_children(const int* parents, int E, int* node_order) {
    long long i = (long long)blockIdx.x * blockDim.x + threadIdx.x;
    long long stride = (long long)gridDim.x * blockDim.x;
    for (; i < E; i += stride) atomicAdd(&node_order[parents[i]], 1);
}

// Deterministic edge ranks (1-based rank among same-parent edges, edge-list
// order). Single block, chunked scan over LDS.
__global__ __launch_bounds__(1024) void tl_edge_rank(const int* parents, int E,
                                                     int* cnt, int* edge_order) {
    __shared__ int par[1024];
    int t = threadIdx.x;
    for (int base = 0; base < E; base += 1024) {
        int i = base + t;
        int p = (i < E) ? parents[i] : -1;
        par[t] = p;
        __syncthreads();
        int lim = imin_i(1024, E - base);
        int rank = 0;
        bool last = true;
        for (int s = 0; s < lim; ++s) {
            if (par[s] == p) {
                if (s < t) rank++;
                else if (s > t) last = false;
            }
        }
        int basec = (i < E) ? cnt[p] : 0;
        __syncthreads();
        if (i < E) {
            edge_order[i] = basec + rank + 1;
            if (last) cnt[p] = basec + rank + 1;
        }
        __syncthreads();
    }
}

__global__ void tl_compact(const int* order, int nitems, int val, int* list,
                           int* counters, int slot) {
    long long i = (long long)blockIdx.x * blockDim.x + threadIdx.x;
    long long stride = (long long)gridDim.x * blockDim.x;
    for (; i < nitems; i += stride) {
        if (order[i] == val) {
            int k = atomicAdd(&counters[slot], 1);
            list[k] = (int)i;
        }
    }
}

// ---------------------------------------------------------------------------
// XW = xb(N x 512 bf16) @ Wall^T(2048 x 512 bf16) + bias.
// Block: 256 threads = 8 waves, tile 32(M) x 128(N), K step 32.
// A tile staged via double-buffered GLOBAL_LOAD_ASYNC_TO_LDS (ASYNCcnt),
// overlapping the global->LDS DMA with WMMA on the other buffer.
// Each wave computes two 16x16 C tiles from one A fragment.
// ---------------------------------------------------------------------------
__global__ __launch_bounds__(256) void tl_gemm_xw(const unsigned short* xb_,
                                                  const unsigned short* Wall_,
                                                  const float* bias, float* C,
                                                  int M) {
    const __bf16* xb = reinterpret_cast<const __bf16*>(xb_);
    const __bf16* W = reinterpret_cast<const __bf16*>(Wall_);
    __shared__ __align__(16) __bf16 As[2][32][40];  // 80B row stride, 16B align

    int tid = threadIdx.x;
    int w = tid >> 5, l = tid & 31;
    int wm = w >> 2, wn = w & 3;
    int lr = l & 15, kh = l >> 4;
    int m0 = blockIdx.y * 32, n0 = blockIdx.x * 128;

    int sr = tid >> 3;           // staging row 0..31
    int sc = (tid & 7) * 4;      // staging col 0..28 (4 bf16 = 8B per thread)
    int gr = imin_i(m0 + sr, M - 1);
    int arow = wm * 16 + lr;
    int col0 = n0 + wn * 16 + lr;
    int col1 = col0 + 64;

    // Prologue: kick off tile 0 DMA.
    async_copy_b64(xb, (unsigned)(((size_t)gr * TL_K + 0 + sc) * 2),
                   &As[0][sr][sc]);

    v8f acc0 = {}, acc1 = {};
    for (int ks = 0; ks < 16; ++ks) {
        int k0 = ks * 32;
        wait_async0();        // this wave's async for tile ks complete
        __syncthreads();      // all waves' asyncs complete; prev reads done
        if (ks < 15) {        // overlap next tile DMA with this tile's math
            async_copy_b64(xb,
                           (unsigned)(((size_t)gr * TL_K + k0 + 32 + sc) * 2),
                           &As[(ks + 1) & 1][sr][sc]);
        }

        v8bf alo = *reinterpret_cast<const v8bf*>(&As[ks & 1][arow][kh * 8]);
        v8bf ahi = *reinterpret_cast<const v8bf*>(&As[ks & 1][arow][16 + kh * 8]);
        v16bf a;
#pragma unroll
        for (int i = 0; i < 8; ++i) { a[i] = alo[i]; a[8 + i] = ahi[i]; }

        acc0 = wmma_bf16(a, load_frag16(W + (size_t)col0 * TL_K, k0, kh), acc0);
        acc1 = wmma_bf16(a, load_frag16(W + (size_t)col1 * TL_K, k0, kh), acc1);
    }
#pragma unroll
    for (int r = 0; r < 8; ++r) {
        int m = m0 + wm * 16 + r + 8 * kh;
        if (m < M) {
            C[(size_t)m * 2048 + col0] = acc0[r] + bias[col0];
            C[(size_t)m * 2048 + col1] = acc1[r] + bias[col1];
        }
    }
}

// ---------------------------------------------------------------------------
// Edge wavefront kernel: for edges with edge_order==n,
//   f = sigmoid(XW_f[p] + h[ch] @ U_f^T + U_f_b); fc[p] = f * c[ch];
//   h_sum[p] += h[ch]  (f32 + bf16 mirror; p unique per wavefront -> no races)
// A fragments come straight from the bf16 mirror hb: no conversion ALU.
// ---------------------------------------------------------------------------
__global__ __launch_bounds__(128) void tl_edge_f(
    const float* h, const unsigned short* hb_, const float* c,
    const unsigned short* Uf_, const float* Ufb, const float* XWall,
    const int* elist, const int* ecnts, int n, const int* parents,
    const int* children, float* fc, float* h_sum, unsigned short* hsumb) {
    const __bf16* Uf = reinterpret_cast<const __bf16*>(Uf_);
    const __bf16* hb = reinterpret_cast<const __bf16*>(hb_);
    int ec = ecnts[n];
    if (ec <= 0) return;
    int w = threadIdx.x >> 5, l = threadIdx.x & 31;
    int lr = l & 15, kh = l >> 4;
    int j0 = (blockIdx.x * 4 + w) * 16;
    int j = j0 + lr;

    for (int mt = blockIdx.y; mt * 16 < ec; mt += gridDim.y) {
        int e_a = elist[imin_i(mt * 16 + lr, ec - 1)];
        const __bf16* arow = hb + (size_t)children[e_a] * TL_K;
        __builtin_prefetch(arow, 0, 0);            // global_prefetch_b8
        __builtin_prefetch(arow + 256, 0, 0);
        v8f acc = {};
#pragma unroll 4
        for (int k0 = 0; k0 < TL_K; k0 += 32) {
            acc = wmma_bf16(load_frag16(arow, k0, kh),
                            load_frag16(Uf + (size_t)j * TL_K, k0, kh), acc);
        }
#pragma unroll
        for (int r = 0; r < 8; ++r) {
            int m = mt * 16 + r + 8 * kh;
            if (m < ec) {
                int e = elist[m];
                int p = parents[e], ch = children[e];
                float fv = acc[r] + XWall[(size_t)p * 2048 + 1536 + j] + Ufb[j];
                fv = sigmoidf_dev(fv);
                fc[(size_t)p * TL_K + j] = fv * c[(size_t)ch * TL_K + j];
                float ns = h_sum[(size_t)p * TL_K + j] + h[(size_t)ch * TL_K + j];
                h_sum[(size_t)p * TL_K + j] = ns;
                hsumb[(size_t)p * TL_K + j] = f32_to_bf16_bits(ns);
            }
        }
    }
}

// ---------------------------------------------------------------------------
// Node wavefront kernel: for nodes with node_order==n,
//   iou = XW_iou[nm] (+ h_sum[nm] @ U_iou^T + U_iou_b if n>0)
//   c[nm] = sigmoid(i)*tanh(u);  o_buf[nm] = sigmoid(o)
// A fragments from the bf16 h_sum mirror; one wave computes i/o/u tiles for
// the same 16-col j range (3 WMMAs / K step, shared A fragment).
// ---------------------------------------------------------------------------
__global__ __launch_bounds__(128) void tl_node_iou(
    const unsigned short* hsumb_, const unsigned short* Uiou_, const float* Uib,
    const float* XWall, const int* nlist, const int* ncnts, int n, int useU,
    float* c, float* o_buf) {
    const __bf16* U = reinterpret_cast<const __bf16*>(Uiou_);
    const __bf16* hsumb = reinterpret_cast<const __bf16*>(hsumb_);
    int nc = ncnts[n];
    if (nc <= 0) return;
    int w = threadIdx.x >> 5, l = threadIdx.x & 31;
    int lr = l & 15, kh = l >> 4;
    int j0 = (blockIdx.x * 4 + w) * 16;
    int j = j0 + lr;

    for (int mt = blockIdx.y; mt * 16 < nc; mt += gridDim.y) {
        v8f aI = {}, aO = {}, aU = {};
        if (useU) {
            int nd_a = nlist[imin_i(mt * 16 + lr, nc - 1)];
            const __bf16* arow = hsumb + (size_t)nd_a * TL_K;
            __builtin_prefetch(arow, 0, 0);
            __builtin_prefetch(arow + 256, 0, 0);
#pragma unroll 2
            for (int k0 = 0; k0 < TL_K; k0 += 32) {
                v16bf a = load_frag16(arow, k0, kh);
                aI = wmma_bf16(a, load_frag16(U + (size_t)j * TL_K, k0, kh), aI);
                aO = wmma_bf16(a, load_frag16(U + (size_t)(512 + j) * TL_K, k0, kh), aO);
                aU = wmma_bf16(a, load_frag16(U + (size_t)(1024 + j) * TL_K, k0, kh), aU);
            }
        }
#pragma unroll
        for (int r = 0; r < 8; ++r) {
            int m = mt * 16 + r + 8 * kh;
            if (m < nc) {
                int nd = nlist[m];
                const float* xw = XWall + (size_t)nd * 2048;
                float vi = xw[j];
                float vo = xw[512 + j];
                float vu = xw[1024 + j];
                if (useU) {
                    vi += aI[r] + Uib[j];
                    vo += aO[r] + Uib[512 + j];
                    vu += aU[r] + Uib[1024 + j];
                }
                float ig = sigmoidf_dev(vi);
                float og = sigmoidf_dev(vo);
                float ug = tanhf(vu);
                c[(size_t)nd * TL_K + j] = ig * ug;       // SET (overwrites)
                o_buf[(size_t)nd * TL_K + j] = og;
            }
        }
    }
}

// c[p] += fc[p] for this wavefront's edges (after the node SET, like reference)
__global__ void tl_c_add(float* c, const float* fc, const int* elist,
                         const int* ecnts, int n, const int* parents) {
    long long ec = ecnts[n];
    long long total = ec * TL_K;
    long long i = (long long)blockIdx.x * blockDim.x + threadIdx.x;
    long long stride = (long long)gridDim.x * blockDim.x;
    for (; i < total; i += stride) {
        int m = (int)(i >> 9), j = (int)(i & 511);
        int p = parents[elist[m]];
        c[(size_t)p * TL_K + j] += fc[(size_t)p * TL_K + j];
    }
}

// h[nm] = o * tanh(c[nm]) (f32 output + bf16 mirror for later WMMA gathers)
__global__ void tl_h_finish(float* h, unsigned short* hb, const float* c,
                            const float* o_buf, const int* nlist,
                            const int* ncnts, int n) {
    long long nc = ncnts[n];
    long long total = nc * TL_K;
    long long i = (long long)blockIdx.x * blockDim.x + threadIdx.x;
    long long stride = (long long)gridDim.x * blockDim.x;
    for (; i < total; i += stride) {
        int m = (int)(i >> 9), j = (int)(i & 511);
        int nd = nlist[m];
        float hv = o_buf[(size_t)nd * TL_K + j] * tanhf(c[(size_t)nd * TL_K + j]);
        h[(size_t)nd * TL_K + j] = hv;
        hb[(size_t)nd * TL_K + j] = f32_to_bf16_bits(hv);
    }
}

// ---------------------------------------------------------------------------
// Host orchestration
// ---------------------------------------------------------------------------
extern "C" void kernel_launch(void* const* d_in, const int* in_sizes, int n_in,
                              void* d_out, int out_size, void* d_ws, size_t ws_size,
                              hipStream_t stream) {
    (void)n_in; (void)out_size; (void)ws_size;
    const float* x       = (const float*)d_in[0];
    const int*   eidx    = (const int*)d_in[1];
    const float* W_iou_w = (const float*)d_in[4];
    const float* W_iou_b = (const float*)d_in[5];
    const float* U_iou_w = (const float*)d_in[6];
    const float* U_iou_b = (const float*)d_in[7];
    const float* W_f_w   = (const float*)d_in[8];
    const float* W_f_b   = (const float*)d_in[9];
    const float* U_f_w   = (const float*)d_in[10];
    const float* U_f_b   = (const float*)d_in[11];

    const long long N = in_sizes[0] / TL_K;
    const long long E = in_sizes[1] / 2;
    const int* parents  = eidx;
    const int* children = eidx + E;
    float* h = (float*)d_out;

    // Workspace carve (256B aligned)
    char* base = (char*)d_ws;
    size_t off = 0;
    auto carve = [&](size_t bytes) -> char* {
        off = (off + 255) & ~(size_t)255;
        char* p = base + off;
        off += bytes;
        return p;
    };
    unsigned short* xb    = (unsigned short*)carve((size_t)N * TL_K * 2);
    unsigned short* Wall  = (unsigned short*)carve((size_t)2048 * TL_K * 2);
    unsigned short* Uioub = (unsigned short*)carve((size_t)1536 * TL_K * 2);
    unsigned short* Ufb16 = (unsigned short*)carve((size_t)512 * TL_K * 2);
    float* bias_all = (float*)carve(2048 * 4);
    float* XWall    = (float*)carve((size_t)N * 2048 * 4);
    float* c        = (float*)carve((size_t)N * TL_K * 4);
    float* h_sum    = (float*)carve((size_t)N * TL_K * 4);
    float* fc       = (float*)carve((size_t)N * TL_K * 4);
    float* o_buf    = (float*)carve((size_t)N * TL_K * 4);
    unsigned short* hb    = (unsigned short*)carve((size_t)N * TL_K * 2);
    unsigned short* hsumb = (unsigned short*)carve((size_t)N * TL_K * 2);
    int* node_order = (int*)carve((size_t)N * 4);
    int* edge_order = (int*)carve((size_t)E * 4);
    int* cnt_tmp    = (int*)carve((size_t)N * 4);
    int* nlist      = (int*)carve((size_t)N * 4);
    int* elist      = (int*)carve((size_t)E * 4);
    int* ncnts      = (int*)carve(TL_ITERS * 4);
    int* ecnts      = (int*)carve(TL_ITERS * 4);

    const dim3 gsb(2048), tb(256);

    // 1) zero state that must start at 0 every call
    tl_zero_u32<<<gsb, tb, 0, stream>>>((unsigned*)h,     N * TL_K);
    tl_zero_u32<<<gsb, tb, 0, stream>>>((unsigned*)c,     N * TL_K);
    tl_zero_u32<<<gsb, tb, 0, stream>>>((unsigned*)h_sum, N * TL_K);
    tl_zero_u32<<<gsb, tb, 0, stream>>>((unsigned*)hb,    N * TL_K / 2);
    tl_zero_u32<<<gsb, tb, 0, stream>>>((unsigned*)node_order, N);
    tl_zero_u32<<<gsb, tb, 0, stream>>>((unsigned*)cnt_tmp,    N);
    tl_zero_u32<<<dim3(1), tb, 0, stream>>>((unsigned*)ncnts, TL_ITERS);
    tl_zero_u32<<<dim3(1), tb, 0, stream>>>((unsigned*)ecnts, TL_ITERS);

    // 2) precision conversion (bf16 weights + activations)
    tl_cvt_bf16<<<gsb, tb, 0, stream>>>(x, xb, N * TL_K);
    tl_cvt_bf16<<<gsb, tb, 0, stream>>>(W_iou_w, Wall, (long long)1536 * TL_K);
    tl_cvt_bf16<<<gsb, tb, 0, stream>>>(W_f_w, Wall + (size_t)1536 * TL_K,
                                        (long long)512 * TL_K);
    tl_cvt_bf16<<<gsb, tb, 0, stream>>>(U_iou_w, Uioub, (long long)1536 * TL_K);
    tl_cvt_bf16<<<gsb, tb, 0, stream>>>(U_f_w, Ufb16, (long long)512 * TL_K);
    tl_copy_f32<<<dim3(8), tb, 0, stream>>>(W_iou_b, bias_all, 1536);
    tl_copy_f32<<<dim3(2), tb, 0, stream>>>(W_f_b, bias_all + 1536, 512);

    // 3) device-side topology
    tl_count_children<<<gsb, tb, 0, stream>>>(parents, (int)E, node_order);
    tl_edge_rank<<<dim3(1), dim3(1024), 0, stream>>>(parents, (int)E, cnt_tmp,
                                                     edge_order);

    // 4) XW = x @ [W_iou|W_f]^T + b  (one big async-staged WMMA GEMM)
    tl_gemm_xw<<<dim3(2048 / 128, (unsigned)((N + 31) / 32)), dim3(256), 0,
                 stream>>>(xb, Wall, bias_all, XWall, (int)N);

    // 5) wavefront 0: leaves (no U term, no U bias)
    tl_compact<<<dim3(512), tb, 0, stream>>>(node_order, (int)N, 0, nlist,
                                             ncnts, 0);
    tl_node_iou<<<dim3(8, 512), dim3(128), 0, stream>>>(
        hsumb, Uioub, U_iou_b, XWall, nlist, ncnts, 0, /*useU=*/0, c, o_buf);
    tl_h_finish<<<gsb, tb, 0, stream>>>(h, hb, c, o_buf, nlist, ncnts, 0);

    // 6) wavefronts 1..TL_ITERS-1 (fixed schedule; empty wavefronts exit fast)
    for (int n = 1; n < TL_ITERS; ++n) {
        tl_compact<<<dim3(512), tb, 0, stream>>>(edge_order, (int)E, n, elist,
                                                 ecnts, n);
        tl_compact<<<dim3(512), tb, 0, stream>>>(node_order, (int)N, n, nlist,
                                                 ncnts, n);
        tl_edge_f<<<dim3(8, 512), dim3(128), 0, stream>>>(
            h, hb, c, Ufb16, U_f_b, XWall, elist, ecnts, n, parents, children,
            fc, h_sum, hsumb);
        tl_node_iou<<<dim3(8, 512), dim3(128), 0, stream>>>(
            hsumb, Uioub, U_iou_b, XWall, nlist, ncnts, n, /*useU=*/1, c, o_buf);
        tl_c_add<<<gsb, tb, 0, stream>>>(c, fc, elist, ecnts, n, parents);
        tl_h_finish<<<gsb, tb, 0, stream>>>(h, hb, c, o_buf, nlist, ncnts, n);
    }
}
